// EdgeConv_74174085202606
// MI455X (gfx1250) — matrix operand order repaired and checked by reading, not compiled, not run on hardware
//
#include <hip/hip_runtime.h>
#include <hip/hip_bf16.h>

typedef _Float16 v8h  __attribute__((ext_vector_type(8)));
typedef _Float16 v16h __attribute__((ext_vector_type(16)));
typedef float    v8f  __attribute__((ext_vector_type(8)));
typedef float    v4f  __attribute__((ext_vector_type(4)));

#define LN_EPS 1e-5f

__device__ __forceinline__ v8f wmma_f16(v16h a, v16h b, v8f c) {
  // D = A(16x32 f16) * B(32x16 f16) + C(16x16 f32)
  return __builtin_amdgcn_wmma_f32_16x16x32_f16(false, a, false, b, (short)0, c,
                                                false, false);
}

__device__ __forceinline__ v8f zero8() {
  v8f z;
#pragma unroll
  for (int i = 0; i < 8; ++i) z[i] = 0.f;
  return z;
}

__device__ __forceinline__ v8h cvt8(const float* __restrict__ p) {
  v4f a = *(const v4f*)p;
  v4f b = *(const v4f*)(p + 4);
  v8h r;
  r[0] = (_Float16)a[0]; r[1] = (_Float16)a[1];
  r[2] = (_Float16)a[2]; r[3] = (_Float16)a[3];
  r[4] = (_Float16)b[0]; r[5] = (_Float16)b[1];
  r[6] = (_Float16)b[2]; r[7] = (_Float16)b[3];
  return r;
}

__device__ __forceinline__ v16h cat16(v8h a, v8h b) {
  return __builtin_shufflevector(a, b, 0, 1, 2, 3, 4, 5, 6, 7,
                                 8, 9, 10, 11, 12, 13, 14, 15);
}

// Stage (din x dout) f32 row-major weights into LDS in WMMA B-fragment order:
// fragment (kb,nt) -> 512 halves, lane = ((k%32)/8%2)*16 + n%16, half idx = k%8 + (k%32)/16*8.
__device__ __forceinline__ void stage_w(const float* __restrict__ W, int din, int dout,
                                        _Float16* __restrict__ sw) {
  int total = din * dout;
  int ntn = dout >> 4;
  for (int idx = threadIdx.x; idx < total; idx += blockDim.x) {
    int k = idx / dout, n = idx - k * dout;
    int kb = k >> 5, r = k & 31;
    int hi = (r >> 3) & 1, i = (r & 7) | ((r >> 4) << 3);
    int nt = n >> 4, nl = n & 15;
    int pos = ((kb * ntn + nt) << 9) + (((hi << 4) | nl) << 4) + i;
    sw[pos] = (_Float16)W[idx];
  }
}

__device__ __forceinline__ v16h load_b(const _Float16* __restrict__ sw, int kb, int nt,
                                       int ntn, int lane) {
  return *(const v16h*)(sw + (((kb * ntn + nt) << 9) + (lane << 4)));
}

// A-fragment from row-major padded (stride 136) LDS activation tile.
// Lane (ha*16+m) holds row m: halves 0..7 = K kb+8ha.. , halves 8..15 = K kb+16+8ha..
__device__ __forceinline__ v16h load_a_lds(const _Float16* __restrict__ act, int m,
                                           int kb, int ha) {
  const _Float16* row = act + m * 136 + kb + 8 * ha;
  v8h c1 = *(const v8h*)(row);
  v8h c2 = *(const v8h*)(row + 16);
  return cat16(c1, c2);
}

// bias + ReLU on C-layout accumulators, scatter into per-wave activation tile
template <int NT>
__device__ __forceinline__ void relu_store_act(const v8f* acc,
                                               const float* __restrict__ bias,
                                               _Float16* __restrict__ act, int lane) {
  int nl = lane & 15, hi = lane >> 4;
#pragma unroll
  for (int nt = 0; nt < NT; ++nt) {
    float bv = bias[nt * 16 + nl];
#pragma unroll
    for (int j = 0; j < 8; ++j) {
      float v = acc[nt][j] + bv;
      v = v > 0.f ? v : 0.f;
      act[(j + 8 * hi) * 136 + nt * 16 + nl] = (_Float16)v;
    }
  }
  asm volatile("s_wait_dscnt 0x0" ::: "memory");  // intra-wave LDS turnaround
}

// bias + LayerNorm over 64 cols of a 16x64 tile held in C-layout (4 N-tiles).
// Row m=j+8*(lane>>4) is spread over one 16-lane half-wave -> shfl_xor width 16.
__device__ __forceinline__ void layernorm16x64(const v8f* acc3,
                                               const float* __restrict__ b3,
                                               const float* __restrict__ g,
                                               const float* __restrict__ be, int lane,
                                               float ov[4][8]) {
  int nl = lane & 15;
  float b3v[4], gv[4], bv[4];
#pragma unroll
  for (int t = 0; t < 4; ++t) {
    int n = t * 16 + nl;
    b3v[t] = b3[n]; gv[t] = g[n]; bv[t] = be[n];
  }
#pragma unroll
  for (int t = 0; t < 4; ++t)
#pragma unroll
    for (int j = 0; j < 8; ++j) ov[t][j] = acc3[t][j] + b3v[t];
#pragma unroll
  for (int j = 0; j < 8; ++j) {
    float s = 0.f, q = 0.f;
#pragma unroll
    for (int t = 0; t < 4; ++t) { float v = ov[t][j]; s += v; q += v * v; }
#pragma unroll
    for (int msk = 1; msk <= 8; msk <<= 1) {
      s += __shfl_xor(s, msk, 16);
      q += __shfl_xor(q, msk, 16);
    }
    float mu  = s * (1.f / 64.f);
    float var = q * (1.f / 64.f) - mu * mu;
    float inv = rsqrtf(var + LN_EPS);
#pragma unroll
    for (int t = 0; t < 4; ++t) ov[t][j] = (ov[t][j] - mu) * inv * gv[t] + bv[t];
  }
}

// ---------------- Kernel 1: per-edge message MLP + scatter-add ----------------
__global__ __launch_bounds__(256) void msg_kernel(
    const float* __restrict__ h, const int* __restrict__ ei,
    const float* __restrict__ ea, const float* __restrict__ W1,
    const float* __restrict__ b1, const float* __restrict__ W2,
    const float* __restrict__ b2, const float* __restrict__ W3,
    const float* __restrict__ b3, const float* __restrict__ g,
    const float* __restrict__ be, float* __restrict__ agg, float* __restrict__ msg,
    int N, int E) {
  // All weights resident in LDS (fragment order) for the whole kernel:
  __shared__ __align__(64) _Float16 sw1[5 * 8 * 512];   // 160x128 -> 40 KB
  __shared__ __align__(64) _Float16 sw2[4 * 8 * 512];   // 128x128 -> 32 KB
  __shared__ __align__(64) _Float16 sw3[4 * 4 * 512];   // 128x64  -> 16 KB
  __shared__ __align__(64) _Float16 sact[8][16 * 136];  // 34 KB (8 waves)

  stage_w(W1, 160, 128, sw1);
  stage_w(W2, 128, 128, sw2);
  stage_w(W3, 128, 64, sw3);
  __syncthreads();

  int lane = threadIdx.x & 31;
  int wave = threadIdx.x >> 5;
  int m = lane & 15, ha = lane >> 4;
  _Float16* act = sact[wave];

  int ntiles = (E + 15) >> 4;
  for (int tile = blockIdx.x * 8 + wave; tile < ntiles; tile += gridDim.x * 8) {
    int base = tile * 16;
    int e0 = base + m;
    int ec = e0 < E ? e0 : E - 1;
    long s = ei[ec], d = ei[E + ec];
    const float* hs  = h + s * 64;
    const float* hd  = h + d * 64;
    const float* eaw = ea + (long)ec * 32;

    // ---- layer 1: (16x160) @ (160x128), A gathered straight from global ----
    v8f acc[8];
#pragma unroll
    for (int t = 0; t < 8; ++t) acc[t] = zero8();
#pragma unroll
    for (int kb = 0; kb < 5; ++kb) {
      const float* p; int off;
      if (kb < 2)      { p = hs;  off = kb * 32; }
      else if (kb < 4) { p = hd;  off = (kb - 2) * 32; }
      else             { p = eaw; off = 0; }
      off += 8 * ha;
      v16h a = cat16(cvt8(p + off), cvt8(p + off + 16));
#pragma unroll
      for (int nt = 0; nt < 8; ++nt)
        acc[nt] = wmma_f16(a, load_b(sw1, kb, nt, 8, lane), acc[nt]);
    }
    relu_store_act<8>(acc, b1, act, lane);

    // ---- layer 2: (16x128) @ (128x128) ----
#pragma unroll
    for (int t = 0; t < 8; ++t) acc[t] = zero8();
#pragma unroll
    for (int kb = 0; kb < 4; ++kb) {
      v16h a = load_a_lds(act, m, kb * 32, ha);
#pragma unroll
      for (int nt = 0; nt < 8; ++nt)
        acc[nt] = wmma_f16(a, load_b(sw2, kb, nt, 8, lane), acc[nt]);
    }
    relu_store_act<8>(acc, b2, act, lane);

    // ---- layer 3: (16x128) @ (128x64) ----
    v8f acc3[4];
#pragma unroll
    for (int t = 0; t < 4; ++t) acc3[t] = zero8();
#pragma unroll
    for (int kb = 0; kb < 4; ++kb) {
      v16h a = load_a_lds(act, m, kb * 32, ha);
#pragma unroll
      for (int nt = 0; nt < 4; ++nt)
        acc3[nt] = wmma_f16(a, load_b(sw3, kb, nt, 4, lane), acc3[nt]);
    }

    float ov[4][8];
    layernorm16x64(acc3, b3, g, be, lane, ov);

    int nl = m;
#pragma unroll
    for (int j = 0; j < 8; ++j) {
      int ej = base + 8 * ha + j;
      if (ej < E) {
        long dj = ei[E + ej];
#pragma unroll
        for (int t = 0; t < 4; ++t) {
          int n = t * 16 + nl;
          float v = ov[t][j];
          msg[(long)ej * 64 + n] = v;
          (void)__hip_atomic_fetch_add(&agg[dj * 64 + n], v, __ATOMIC_RELAXED,
                                       __HIP_MEMORY_SCOPE_AGENT);
        }
      }
    }
  }
}

// ---------------- Kernel 2: per-node update MLP + residual ----------------
__global__ __launch_bounds__(256) void upd_kernel(
    const float* __restrict__ h, const float* __restrict__ W1,
    const float* __restrict__ b1, const float* __restrict__ W2,
    const float* __restrict__ b2, const float* __restrict__ W3,
    const float* __restrict__ b3, const float* __restrict__ g,
    const float* __restrict__ be, float* __restrict__ hnew /* agg in, h_new out */,
    int N) {
  __shared__ __align__(64) _Float16 sw1[4 * 8 * 512];   // 128x128 -> 32 KB
  __shared__ __align__(64) _Float16 sw2[4 * 8 * 512];   // 128x128 -> 32 KB
  __shared__ __align__(64) _Float16 sw3[4 * 4 * 512];   // 128x64  -> 16 KB
  __shared__ __align__(64) _Float16 sact[8][16 * 136];

  stage_w(W1, 128, 128, sw1);
  stage_w(W2, 128, 128, sw2);
  stage_w(W3, 128, 64, sw3);
  __syncthreads();

  int lane = threadIdx.x & 31;
  int wave = threadIdx.x >> 5;
  int m = lane & 15, ha = lane >> 4;
  _Float16* act = sact[wave];

  int ntiles = (N + 15) >> 4;
  for (int tile = blockIdx.x * 8 + wave; tile < ntiles; tile += gridDim.x * 8) {
    int base = tile * 16;
    int n0 = base + m;
    int nc = n0 < N ? n0 : N - 1;
    const float* hr = h + (long)nc * 64;
    const float* ar = hnew + (long)nc * 64;  // agg (only this wave owns its rows)

    // ---- layer 1: (16x128 concat[h,agg]) @ (128x128) ----
    v8f acc[8];
#pragma unroll
    for (int t = 0; t < 8; ++t) acc[t] = zero8();
#pragma unroll
    for (int kb = 0; kb < 4; ++kb) {
      const float* p = (kb < 2) ? hr : ar;
      int off = (kb & 1) * 32 + 8 * ha;
      v16h a = cat16(cvt8(p + off), cvt8(p + off + 16));
#pragma unroll
      for (int nt = 0; nt < 8; ++nt)
        acc[nt] = wmma_f16(a, load_b(sw1, kb, nt, 8, lane), acc[nt]);
    }
    relu_store_act<8>(acc, b1, act, lane);

    // ---- layer 2 ----
#pragma unroll
    for (int t = 0; t < 8; ++t) acc[t] = zero8();
#pragma unroll
    for (int kb = 0; kb < 4; ++kb) {
      v16h a = load_a_lds(act, m, kb * 32, ha);
#pragma unroll
      for (int nt = 0; nt < 8; ++nt)
        acc[nt] = wmma_f16(a, load_b(sw2, kb, nt, 8, lane), acc[nt]);
    }
    relu_store_act<8>(acc, b2, act, lane);

    // ---- layer 3 ----
    v8f acc3[4];
#pragma unroll
    for (int t = 0; t < 4; ++t) acc3[t] = zero8();
#pragma unroll
    for (int kb = 0; kb < 4; ++kb) {
      v16h a = load_a_lds(act, m, kb * 32, ha);
#pragma unroll
      for (int nt = 0; nt < 4; ++nt)
        acc3[nt] = wmma_f16(a, load_b(sw3, kb, nt, 4, lane), acc3[nt]);
    }

    float ov[4][8];
    layernorm16x64(acc3, b3, g, be, lane, ov);

    int nl = m;
#pragma unroll
    for (int j = 0; j < 8; ++j) {
      int nj = base + 8 * ha + j;
      if (nj < N) {
#pragma unroll
        for (int t = 0; t < 4; ++t) {
          int n = t * 16 + nl;
          hnew[(long)nj * 64 + n] = ov[t][j] + h[(long)nj * 64 + n];
        }
      }
    }
  }
}

extern "C" void kernel_launch(void* const* d_in, const int* in_sizes, int n_in,
                              void* d_out, int out_size, void* d_ws, size_t ws_size,
                              hipStream_t stream) {
  const float* h   = (const float*)d_in[0];
  const int*   ei  = (const int*)d_in[1];
  const float* ea  = (const float*)d_in[2];
  const float* mW1 = (const float*)d_in[3];
  const float* mb1 = (const float*)d_in[4];
  const float* mW2 = (const float*)d_in[5];
  const float* mb2 = (const float*)d_in[6];
  const float* mW3 = (const float*)d_in[7];
  const float* mb3 = (const float*)d_in[8];
  const float* mg  = (const float*)d_in[9];
  const float* mbe = (const float*)d_in[10];
  const float* uW1 = (const float*)d_in[11];
  const float* ub1 = (const float*)d_in[12];
  const float* uW2 = (const float*)d_in[13];
  const float* ub2 = (const float*)d_in[14];
  const float* uW3 = (const float*)d_in[15];
  const float* ub3 = (const float*)d_in[16];
  const float* ug  = (const float*)d_in[17];
  const float* ube = (const float*)d_in[18];

  int N = in_sizes[0] / 64;
  int E = in_sizes[2] / 32;

  float* hnew = (float*)d_out;                    // doubles as agg accumulator
  float* msg  = (float*)d_out + (size_t)N * 64;

  hipMemsetAsync(hnew, 0, (size_t)N * 64 * sizeof(float), stream);

  int nt1 = (E + 15) / 16;
  int blocks1 = (nt1 + 7) / 8;
  if (blocks1 > 800) blocks1 = 800;               // ~4-5 tiles/wave, amortize staging
  msg_kernel<<<blocks1, 256, 0, stream>>>(h, ei, ea, mW1, mb1, mW2, mb2, mW3, mb3,
                                          mg, mbe, hnew, msg, N, E);

  int nt2 = (N + 15) / 16;
  int blocks2 = (nt2 + 7) / 8;
  if (blocks2 > 800) blocks2 = 800;
  upd_kernel<<<blocks2, 256, 0, stream>>>(h, uW1, ub1, uW2, ub2, uW3, ub3, ug, ube,
                                          hnew, N);
}
